// GRU_29824252903876
// MI455X (gfx1250) — matrix operand order repaired and checked by reading, not compiled, not run on hardware
//
#include <hip/hip_runtime.h>
#include <hip/hip_bf16.h>

#define B_ 64
#define S_ 1024
#define I_ 1024
#define H_ 1024
#define NWG 32

typedef __attribute__((ext_vector_type(16))) __bf16 v16bf;
typedef __attribute__((ext_vector_type(2)))  __bf16 v2bf;
typedef __attribute__((ext_vector_type(8)))  float  v8f;

union Frag16 { v16bf v; uint4 u[2]; };

// Pointer types for the async global->LDS DMA builtin:
// signature is (v4i addrspace(1)*, v4i addrspace(3)*, imm offset, imm cpol).
typedef int v4i_ __attribute__((vector_size(16)));
typedef __attribute__((address_space(1))) v4i_ GV4;
typedef __attribute__((address_space(3))) v4i_ LV4;

#if __has_builtin(__builtin_amdgcn_global_load_async_to_lds_b128)
#define HAVE_ASYNC_LDS 1
#define ASYNC_B128(g, l) \
  __builtin_amdgcn_global_load_async_to_lds_b128((GV4*)(g), (LV4*)(l), 0, 0)
#if __has_builtin(__builtin_amdgcn_s_wait_asynccnt)
#define WAIT_ASYNC(n) __builtin_amdgcn_s_wait_asynccnt(n)
#else
#define WAIT_ASYNC(n) asm volatile("s_wait_asynccnt %0" ::"i"(n) : "memory")
#endif
#else
#define HAVE_ASYNC_LDS 0
#define WAIT_ASYNC(n)
#endif

// ---------------------------------------------------------------------------
// Tensor Data Mover (TDM): one-instruction 2D tile DMA global->LDS.
// Descriptor bitfields per CDNA5 ISA ch.8 (D# groups 0..3).
// ---------------------------------------------------------------------------
#if __has_builtin(__builtin_amdgcn_tensor_load_to_lds)
#define HAVE_TDM 1
typedef __attribute__((ext_vector_type(4))) unsigned u32x4;
typedef __attribute__((ext_vector_type(8))) int      i32x8;
typedef __attribute__((ext_vector_type(4))) int      i32x4;

#if __has_builtin(__builtin_amdgcn_s_wait_tensorcnt)
#define WAIT_TENSOR(n) __builtin_amdgcn_s_wait_tensorcnt(n)
#else
#define WAIT_TENSOR(n) asm volatile("s_wait_tensorcnt %0" ::"i"(n) : "memory")
#endif

__device__ inline void tdm_load_2d_bf16(const unsigned short* gsrc,
                                        void* ldst, unsigned tile_w,
                                        unsigned tile_h, unsigned stride_elems) {
  unsigned long long ga = (unsigned long long)(uintptr_t)gsrc;
  const unsigned DB = 0x100000u;  // generous tensor dims: no OOB clipping
  u32x4 g0;
  g0[0] = 1u;                                                 // count=1
  g0[1] = (unsigned)(uintptr_t)ldst;                          // lds_addr (bytes)
  g0[2] = (unsigned)ga;                                       // global_addr lo
  g0[3] = (unsigned)((ga >> 32) & 0x01FFFFFFu) | (2u << 30);  // addr hi | type=2
  i32x8 g1;
  g1[0] = (int)(1u << 16);                                    // data_size=2B
  g1[1] = (int)((DB & 0xFFFFu) << 16);                        // tensor_dim0 lo
  g1[2] = (int)(((DB >> 16) & 0xFFFFu) | ((DB & 0xFFFFu) << 16)); // td0 hi|td1 lo
  g1[3] = (int)(((DB >> 16) & 0xFFFFu) | (tile_w << 16));     // td1 hi|tile_dim0
  g1[4] = (int)(tile_h & 0xFFFFu);                            // tile_dim1
  g1[5] = (int)stride_elems;                                  // dim0_stride lo
  g1[6] = 0;
  g1[7] = 0;
  i32x4 z4;
#pragma unroll
  for (int i = 0; i < 4; ++i) z4[i] = 0;
#if __clang_major__ >= 23
  i32x8 z8;
#pragma unroll
  for (int i = 0; i < 8; ++i) z8[i] = 0;
  __builtin_amdgcn_tensor_load_to_lds(g0, g1, z4, z4, z8, 0);
#else
  __builtin_amdgcn_tensor_load_to_lds(g0, g1, z4, z4, 0);
#endif
}
#else
#define HAVE_TDM 0
#endif

__device__ inline unsigned short f2bf(float f) {
  return __builtin_bit_cast(unsigned short, (__bf16)f);  // native RNE cvt
}
__device__ inline float bf2f(unsigned short h) {
  return __uint_as_float(((unsigned)h) << 16);
}
__device__ inline unsigned pack2(float lo, float hi) {
  v2bf p;
  p[0] = (__bf16)lo;
  p[1] = (__bf16)hi;
  return __builtin_bit_cast(unsigned, p);
}
__device__ inline v8f zero8() {
  v8f z;
#pragma unroll
  for (int i = 0; i < 8; ++i) z[i] = 0.0f;
  return z;
}

// ---------------------------------------------------------------------------
// Phase 0: fp32 -> bf16 conversion (weights + h0)
// ---------------------------------------------------------------------------
__global__ void cvt_kernel(const float* __restrict__ src,
                           unsigned short* __restrict__ dst, int n) {
  int i = blockIdx.x * blockDim.x + threadIdx.x;
  if (i < n) dst[i] = f2bf(src[i]);
}

// ---------------------------------------------------------------------------
// Phase 1: input projections. x_g = X @ W_g^T + b_g  (bf16 out)
// Grid: (BS/128, H/64, 3 gates). Block: 256 threads = 8 waves.
// ---------------------------------------------------------------------------
__global__ __launch_bounds__(256) void xproj_kernel(
    const float* __restrict__ X,               // [BS, I] fp32
    const unsigned short* __restrict__ Wbf,    // [3, H, I] bf16
    const float* __restrict__ br,
    const float* __restrict__ bz,
    const float* __restrict__ bn,
    unsigned short* __restrict__ Xout)         // [3, BS, H] bf16
{
  const int g  = blockIdx.z;
  const int m0 = blockIdx.x * 128;
  const int n0 = blockIdx.y * 64;
  const unsigned short* W = Wbf + (size_t)g * H_ * I_;
  unsigned short* O = Xout + (size_t)g * (size_t)(B_ * S_) * H_;
  const float* bias = (g == 0) ? br : (g == 1) ? bz : bn;

  __shared__ unsigned short Asb[128 * 32];  // A tile, bf16, row-major, ld=32
  __shared__ unsigned short Bsb[64 * 32];   // B tile (weights), ld=32

  const int t    = threadIdx.x;
  const int lane = t & 31;
  const int wave = t >> 5;
  const int wm   = (wave >> 1) * 32;
  const int wn   = (wave & 1) * 32;

  v8f c[2][2];
#pragma unroll
  for (int i = 0; i < 2; ++i)
#pragma unroll
    for (int j = 0; j < 2; ++j) c[i][j] = zero8();

  for (int k0 = 0; k0 < I_; k0 += 32) {
    __syncthreads();
    if (t < 128) {  // B: 64x32 bf16 -> LDS (async, overlaps A conversion)
      int n  = t >> 1;
      int kc = (t & 1) * 16;
      const unsigned short* src = W + (size_t)(n0 + n) * I_ + k0 + kc;
      unsigned short* dst = &Bsb[n * 32 + kc];
#if HAVE_ASYNC_LDS
      ASYNC_B128(src, dst);
      ASYNC_B128(src + 8, dst + 8);
#else
      uint4 q0 = ((const uint4*)src)[0];
      uint4 q1 = ((const uint4*)src)[1];
      *(uint4*)dst       = q0;
      *(uint4*)(dst + 8) = q1;
#endif
    }
    {  // A: 128x32 fp32 -> bf16 into LDS (16 floats per thread)
      int row = t >> 1;
      int kc  = (t & 1) * 16;
      const float* src = X + (size_t)(m0 + row) * I_ + k0 + kc;
      if (k0 + 32 < I_) __builtin_prefetch(src + 32, 0, 1);
      float4 f0 = ((const float4*)src)[0];
      float4 f1 = ((const float4*)src)[1];
      float4 f2 = ((const float4*)src)[2];
      float4 f3 = ((const float4*)src)[3];
      uint4 p0, p1;
      p0.x = pack2(f0.x, f0.y); p0.y = pack2(f0.z, f0.w);
      p0.z = pack2(f1.x, f1.y); p0.w = pack2(f1.z, f1.w);
      p1.x = pack2(f2.x, f2.y); p1.y = pack2(f2.z, f2.w);
      p1.z = pack2(f3.x, f3.y); p1.w = pack2(f3.z, f3.w);
      *(uint4*)&Asb[row * 32 + kc]     = p0;
      *(uint4*)&Asb[row * 32 + kc + 8] = p1;
    }
#if HAVE_ASYNC_LDS
    if (t < 128) { WAIT_ASYNC(0); }
#endif
    __syncthreads();

    Frag16 a[2], b[2];
#pragma unroll
    for (int im = 0; im < 2; ++im) {
      int m  = wm + im * 16 + (lane & 15);
      int kb = (lane >> 4) * 8;
      a[im].u[0] = *(const uint4*)&Asb[m * 32 + kb];
      a[im].u[1] = *(const uint4*)&Asb[m * 32 + kb + 16];
    }
#pragma unroll
    for (int jn = 0; jn < 2; ++jn) {
      int n  = wn + jn * 16 + (lane & 15);
      int kb = (lane >> 4) * 8;
      b[jn].u[0] = *(const uint4*)&Bsb[n * 32 + kb];
      b[jn].u[1] = *(const uint4*)&Bsb[n * 32 + kb + 16];
    }
#pragma unroll
    for (int im = 0; im < 2; ++im)
#pragma unroll
      for (int jn = 0; jn < 2; ++jn)
        c[im][jn] = __builtin_amdgcn_wmma_f32_16x16x32_bf16(
            false, a[im].v, false, b[jn].v, (short)0, c[im][jn], false, false);
  }

  // Epilogue: + bias, fp32 -> bf16, store
#pragma unroll
  for (int jn = 0; jn < 2; ++jn) {
    int n    = n0 + wn + jn * 16 + (lane & 15);
    float bv = bias[n];
#pragma unroll
    for (int im = 0; im < 2; ++im) {
#pragma unroll
      for (int v = 0; v < 8; ++v) {
        int m = m0 + wm + im * 16 + v + ((lane >> 4) * 8);
        O[(size_t)m * H_ + n] = f2bf(c[im][jn][v] + bv);
      }
    }
  }
}

// ---------------------------------------------------------------------------
// h_prev chunk staging for the scan kernel: 64x32 bf16 (128 threads).
// ---------------------------------------------------------------------------
__device__ inline void stage_h_chunk(const unsigned short* __restrict__ hprev,
                                     unsigned short* __restrict__ buf,
                                     int k0, int t) {
  int row = t >> 1;
  int kc  = (t & 1) * 16;
  const unsigned short* src = hprev + (size_t)row * H_ + k0 + kc;
  unsigned short* dst = buf + row * 32 + kc;
#if HAVE_ASYNC_LDS
  ASYNC_B128(src, dst);
  ASYNC_B128(src + 8, dst + 8);
#else
  uint4 q0 = ((const uint4*)src)[0];
  uint4 q1 = ((const uint4*)src)[1];
  *(uint4*)dst       = q0;
  *(uint4*)(dst + 8) = q1;
#endif
}

// ---------------------------------------------------------------------------
// Phase 2: persistent recurrent scan. 32 WGs x 192 threads (6 waves).
// WG owns 32 output columns; LDS holds its bf16 slice of w_hr/w_hz/w_hn
// (192 KB, needs CDNA5's 320 KB WGP LDS) for all 1024 steps; preloaded
// by the Tensor Data Mover when available.
// ---------------------------------------------------------------------------
__global__ __launch_bounds__(192) void scan_kernel(
    const unsigned short* __restrict__ Whbf,   // [3, H, H] bf16 (r,z,n)
    const float* __restrict__ bhr,
    const float* __restrict__ bhz,
    const float* __restrict__ bhn,
    const unsigned short* __restrict__ Xg,     // [3, BS, H] bf16
    unsigned short* __restrict__ hbuf,         // [2, B, H] bf16
    float* __restrict__ out,                   // [B*S*H + B*H] fp32
    unsigned* __restrict__ ctr)
{
  extern __shared__ char smem[];
  unsigned short* Wl = (unsigned short*)smem;                        // [96][1024]
  unsigned short* Al = (unsigned short*)(smem + 3 * 32 * 1024 * 2);  // [2][64][32]
  float* Gacc = (float*)(smem + 3 * 32 * 1024 * 2 + 2 * 64 * 32 * 2); // [3][64][32]

  const int wg   = blockIdx.x;
  const int n0   = wg * 32;
  const int t    = threadIdx.x;
  const int lane = t & 31;
  const int wave = t >> 5;      // 0..5
  const int gate = wave >> 1;   // 0..2
  const int nt   = wave & 1;    // 0..1

  // Preload this WG's weight slice: 3 x (32 rows x 1024 bf16) = 192 KB (once).
#if HAVE_TDM
  if (wave == 0) {  // one TDM 2D-tile descriptor per gate; TENSORcnt tracked
#pragma unroll
    for (int g = 0; g < 3; ++g) {
      tdm_load_2d_bf16(Whbf + ((size_t)g * H_ + n0) * H_,
                       Wl + (size_t)g * 32 * 1024,
                       /*tile_w=*/1024, /*tile_h=*/32, /*stride=*/H_);
    }
    WAIT_TENSOR(0);
  }
#else
  for (int i = t; i < 96 * 128; i += 192) {          // uint4 granules
    int rowIdx = i >> 7;                              // 0..95
    int chunk  = (i & 127) * 8;
    int g = rowIdx >> 5, r = rowIdx & 31;
    const unsigned short* src =
        Whbf + ((size_t)g * H_ + (n0 + r)) * H_ + chunk;
    unsigned short* dst = &Wl[(size_t)rowIdx * 1024 + chunk];
#if HAVE_ASYNC_LDS
    ASYNC_B128(src, dst);
#else
    *(uint4*)dst = *(const uint4*)src;
#endif
  }
  WAIT_ASYNC(0);
#endif
  __syncthreads();

  for (int s = 0; s < S_; ++s) {
    const unsigned short* hprev = hbuf + (size_t)(s & 1) * (B_ * H_);
    unsigned short*       hnext = hbuf + (size_t)((s + 1) & 1) * (B_ * H_);

    v8f acc[4];
#pragma unroll
    for (int mt = 0; mt < 4; ++mt) acc[mt] = zero8();

    if (t < 128) stage_h_chunk(hprev, Al, 0, t);  // chunk 0 into buf 0

    for (int kc = 0; kc < 32; ++kc) {
      const unsigned short* cur = Al + (kc & 1) * (64 * 32);
      if (t < 128) {
        if (kc + 1 < 32) {  // prefetch next chunk into other buffer
          stage_h_chunk(hprev, Al + ((kc + 1) & 1) * (64 * 32),
                        (kc + 1) * 32, t);
          WAIT_ASYNC(2);    // chunk kc landed; kc+1 still in flight
        } else {
          WAIT_ASYNC(0);
        }
      }
      __syncthreads();

      Frag16 bfrag;
      {
        int n  = nt * 16 + (lane & 15);
        int kb = kc * 32 + (lane >> 4) * 8;
        const unsigned short* wrow = Wl + (size_t)(gate * 32 + n) * 1024;
        bfrag.u[0] = *(const uint4*)&wrow[kb];
        bfrag.u[1] = *(const uint4*)&wrow[kb + 16];
      }
#pragma unroll
      for (int mt = 0; mt < 4; ++mt) {
        Frag16 afrag;
        int m  = mt * 16 + (lane & 15);
        int kb = (lane >> 4) * 8;
        afrag.u[0] = *(const uint4*)&cur[m * 32 + kb];
        afrag.u[1] = *(const uint4*)&cur[m * 32 + kb + 16];
        acc[mt] = __builtin_amdgcn_wmma_f32_16x16x32_bf16(
            false, afrag.v, false, bfrag.v, (short)0, acc[mt], false, false);
      }
      __syncthreads();
    }

    // Exchange gate pre-activations through LDS.
#pragma unroll
    for (int mt = 0; mt < 4; ++mt) {
      int n = nt * 16 + (lane & 15);
#pragma unroll
      for (int v = 0; v < 8; ++v) {
        int m = mt * 16 + v + ((lane >> 4) * 8);
        Gacc[((size_t)gate * 64 + m) * 32 + n] = acc[mt][v];
      }
    }
    __syncthreads();

    // Fused gate math + h update for the 64x32 slice.
    for (int e = t; e < 64 * 32; e += 192) {
      int m  = e >> 5;       // batch
      int nl = e & 31;
      int n  = n0 + nl;
      size_t xoff = ((size_t)m * S_ + s) * H_ + n;
      float xr = bf2f(Xg[xoff]);
      float xz = bf2f(Xg[(size_t)(B_ * S_) * H_ + xoff]);
      float xn = bf2f(Xg[2 * (size_t)(B_ * S_) * H_ + xoff]);
      float ar = Gacc[(0 * 64 + m) * 32 + nl];
      float az = Gacc[(1 * 64 + m) * 32 + nl];
      float an = Gacc[(2 * 64 + m) * 32 + nl];
      float r  = 1.0f / (1.0f + __expf(-(xr + ar + bhr[n])));
      float z  = 1.0f / (1.0f + __expf(-(xz + az + bhz[n])));
      float nn = tanhf(xn + r * (an + bhn[n]));
      float hp = bf2f(hprev[(size_t)m * H_ + n]);
      float hv = (1.0f - z) * nn + z * hp;
      out[((size_t)m * S_ + s) * H_ + n] = hv;
      hnext[(size_t)m * H_ + n] = f2bf(hv);
      if (s == S_ - 1) out[(size_t)B_ * S_ * H_ + (size_t)m * H_ + n] = hv;
    }

    // Device-wide barrier (monotonic counter, zeroed each launch).
    __syncthreads();
    if (t == 0) {
      __hip_atomic_fetch_add(ctr, 1u, __ATOMIC_ACQ_REL, __HIP_MEMORY_SCOPE_AGENT);
      unsigned target = (unsigned)(NWG * (s + 1));
      while (__hip_atomic_load(ctr, __ATOMIC_ACQUIRE, __HIP_MEMORY_SCOPE_AGENT) < target)
        __builtin_amdgcn_s_sleep(2);
    }
    __syncthreads();
  }
}

// ---------------------------------------------------------------------------
extern "C" void kernel_launch(void* const* d_in, const int* in_sizes, int n_in,
                              void* d_out, int out_size, void* d_ws, size_t ws_size,
                              hipStream_t stream) {
  (void)in_sizes; (void)n_in; (void)out_size; (void)ws_size;

  const float* X    = (const float*)d_in[0];
  const float* h0   = (const float*)d_in[1];
  const float* w_ir = (const float*)d_in[2];
  const float* w_iz = (const float*)d_in[3];
  const float* w_in = (const float*)d_in[4];
  const float* b_ir = (const float*)d_in[5];
  const float* b_iz = (const float*)d_in[6];
  const float* b_in = (const float*)d_in[7];
  const float* w_hr = (const float*)d_in[8];
  const float* w_hz = (const float*)d_in[9];
  const float* w_hn = (const float*)d_in[10];
  const float* b_hr = (const float*)d_in[11];
  const float* b_hz = (const float*)d_in[12];
  const float* b_hn = (const float*)d_in[13];

  char* ws = (char*)d_ws;
  size_t off = 0;
  unsigned* ctr = (unsigned*)ws;                              off += 256;
  unsigned short* wi_bf = (unsigned short*)(ws + off);        off += (size_t)3 * H_ * I_ * 2;
  unsigned short* wh_bf = (unsigned short*)(ws + off);        off += (size_t)3 * H_ * H_ * 2;
  unsigned short* hbuf  = (unsigned short*)(ws + off);        off += (size_t)2 * B_ * H_ * 2;
  unsigned short* xg_bf = (unsigned short*)(ws + off);        off += (size_t)3 * B_ * S_ * H_ * 2;

  (void)hipMemsetAsync(ctr, 0, 256, stream);

  const int nW = H_ * I_;
  const int cb = 256;
  cvt_kernel<<<(nW + cb - 1) / cb, cb, 0, stream>>>(w_ir, wi_bf + (size_t)0 * nW, nW);
  cvt_kernel<<<(nW + cb - 1) / cb, cb, 0, stream>>>(w_iz, wi_bf + (size_t)1 * nW, nW);
  cvt_kernel<<<(nW + cb - 1) / cb, cb, 0, stream>>>(w_in, wi_bf + (size_t)2 * nW, nW);
  cvt_kernel<<<(nW + cb - 1) / cb, cb, 0, stream>>>(w_hr, wh_bf + (size_t)0 * nW, nW);
  cvt_kernel<<<(nW + cb - 1) / cb, cb, 0, stream>>>(w_hz, wh_bf + (size_t)1 * nW, nW);
  cvt_kernel<<<(nW + cb - 1) / cb, cb, 0, stream>>>(w_hn, wh_bf + (size_t)2 * nW, nW);
  cvt_kernel<<<(B_ * H_ + cb - 1) / cb, cb, 0, stream>>>(h0, hbuf, B_ * H_);

  dim3 g1(B_ * S_ / 128, H_ / 64, 3);
  xproj_kernel<<<g1, 256, 0, stream>>>(X, wi_bf, b_ir, b_iz, b_in, xg_bf);

  const int scan_lds = 3 * 32 * 1024 * 2 + 2 * 64 * 32 * 2 + 3 * 64 * 32 * 4; // 224 KB
  (void)hipFuncSetAttribute((const void*)scan_kernel,
                            hipFuncAttributeMaxDynamicSharedMemorySize, scan_lds);
  scan_kernel<<<NWG, 192, scan_lds, stream>>>(wh_bf, b_hr, b_hz, b_hn, xg_bf,
                                              hbuf, (float*)d_out, ctr);
}
